// TransformerEncoder_81114752352883
// MI455X (gfx1250) — compile-verified, matrix operand order checked by reading, not run on hardware
//
#include <hip/hip_runtime.h>
#include <hip/hip_bf16.h>
#include <cstdint>
#include <cstddef>

// ---------------- problem constants (fixed by the reference) ----------------
#define NLAYER   4
#define NHEAD    8
#define DMODEL   512
#define DFF      2048
#define HEADDIM  64
#define BATCH    2
#define SEQ      2048
#define ROWS     (BATCH * SEQ)      // 4096 token rows
#define LNEPS    1e-5f
#define INV_SCALE 0.125f            // 1/sqrt(64)

typedef __attribute__((ext_vector_type(16))) _Float16 v16h;
typedef __attribute__((ext_vector_type(8)))  _Float16 v8h;
typedef __attribute__((ext_vector_type(8)))  float    v8f;

union FragU { v16h v; v8h h[2]; };

// ---------------- WMMA helpers ----------------
__device__ __forceinline__ v8f wmma_f16(v16h a, v16h b, v8f c) {
  // D = A(16x32 f16) * B(32x16 f16) + C(16x16 f32)
  return __builtin_amdgcn_wmma_f32_16x16x32_f16(
      /*neg_a=*/false, a, /*neg_b=*/false, b,
      /*c_mod=*/(short)0, c, /*reuse_a=*/false, /*reuse_b=*/false);
}

// LDS tile: rows x 32 halfs, padded row stride 40 halfs (80B, 16B aligned).
// Fragment layout per CDNA5 ISA (16-bit A/B 16x32):
//   lanes 0-15 : row = lane,    K = {0..7, 16..23}
//   lanes 16-31: row = lane-16, K = {8..15, 24..31}
#define LDS_STRIDE 40
__device__ __forceinline__ v16h ld_frag(const _Float16* __restrict__ s,
                                        int row0, int lane) {
  int r  = row0 + (lane & 15);
  int lo = (lane >> 4) << 3;     // 0 or 8 halfs
  const _Float16* p = s + r * LDS_STRIDE + lo;
  FragU f;
  f.h[0] = *(const v8h*)(p);          // K block 0..7  (or 8..15)
  f.h[1] = *(const v8h*)(p + 16);     // K block 16..23 (or 24..31)
  return f.v;
}

// Issue one K-stage (64xBN block, 32-wide K) of async global->LDS copies.
// 16B per lane per instruction; 2 + BN/32 instructions per thread.
// Branch-free: A-loop and B-loop have compile-time trip counts, and tid is
// masked so the compiler sees uniform control flow (no EXEC save/restore).
template <int BN>
__device__ __forceinline__ void stage_async(
    const _Float16* __restrict__ Ab, const _Float16* __restrict__ Wb,
    int lda, int ldw, int bm, int bn, int kb,
    unsigned ldsA, unsigned ldsB, int tid) {
  const int t = tid & 127;
#pragma unroll
  for (int i = 0; i < 2; ++i) {          // A tile: 64 rows * 4 chunks = 256
    int c = t + (i << 7);
    int r = c >> 2, cc = (c & 3) << 3;
    const void* g = &Ab[(size_t)(bm + r) * lda + kb + cc];
    unsigned l = ldsA + (unsigned)((r * LDS_STRIDE + cc) * 2);
    asm volatile("global_load_async_to_lds_b128 %0, %1, off"
                 :: "v"(l), "v"(g) : "memory");
  }
#pragma unroll
  for (int i = 0; i < (BN >> 5); ++i) {  // B tile: BN rows * 4 chunks
    int c = t + (i << 7);
    int r = c >> 2, cc = (c & 3) << 3;
    const void* g = &Wb[(size_t)(bn + r) * ldw + kb + cc];
    unsigned l = ldsB + (unsigned)((r * LDS_STRIDE + cc) * 2);
    asm volatile("global_load_async_to_lds_b128 %0, %1, off"
                 :: "v"(l), "v"(g) : "memory");
  }
}

// ---------------- GEMM epilogue modes ----------------
enum {
  EP_SCORES = 0,  // f16 out, *INV_SCALE, batched over (b,h), row-major [T,T]
  EP_QKHEAD = 1,  // f16 out scattered to [B,H,T,64], +bias
  EP_VTRANS = 2,  // f16 out scattered to [B,H,64,T], +bias
  EP_ATTNO  = 3,  // f16 out scattered to [B,T,DMODEL], batched over (b,h)
  EP_F16BR  = 4,  // f16 out row-major, +bias, ReLU (FFN-1)
  EP_RESID  = 5,  // f32 out row-major, +bias, += residual in place
};

// C = A[M,K] (f16 row-major, lda) x W[N,K]^T (f16 row-major, ldw)  (+epilogue)
// Block: 128 threads = 4 waves. Block tile 64xBN, wave tile 32x(BN/2).
// Double-buffered LDS fed by async copies (ASYNCcnt-managed).
template <int MODE, int BN>
__global__ __launch_bounds__(128)
void gemm_wmma(const _Float16* __restrict__ A, const _Float16* __restrict__ W,
               const float* __restrict__ bias,
               _Float16* __restrict__ outh, float* __restrict__ outf,
               int M, int N, int K, int lda, int ldw, int ldo,
               long long strideA, long long strideW, long long strideO) {
  __shared__ _Float16 sA[2][64 * LDS_STRIDE];
  __shared__ _Float16 sB[2][BN * LDS_STRIDE];

  constexpr int WN = BN / 2;        // wave tile N
  constexpr int NF = WN / 16;       // B fragments per wave

  const int batch = blockIdx.z;
  const _Float16* __restrict__ Ab = A + (long long)batch * strideA;
  const _Float16* __restrict__ Wb = W + (long long)batch * strideW;

  const int tid  = threadIdx.x;
  const int lane = tid & 31;
  const int wid  = tid >> 5;              // 0..3
  const int wm   = (wid >> 1) << 5;       // 0 or 32
  const int wn   = (wid & 1) * WN;
  const int bm   = blockIdx.y << 6;
  const int bn   = blockIdx.x * BN;

  const unsigned baseA = (unsigned)(uintptr_t)(&sA[0][0]);
  const unsigned baseB = (unsigned)(uintptr_t)(&sB[0][0]);
  constexpr unsigned bufA = 64 * LDS_STRIDE * 2;   // bytes per buffer
  constexpr unsigned bufB = BN * LDS_STRIDE * 2;

  v8f c[2][NF] = {};

  auto compute_stage = [&](int cur) {
    const _Float16* tA = &sA[cur][0];
    const _Float16* tB = &sB[cur][0];
    v16h a0 = ld_frag(tA, wm,      lane);
    v16h a1 = ld_frag(tA, wm + 16, lane);
#pragma unroll
    for (int in = 0; in < NF; ++in) {
      v16h b = ld_frag(tB, wn + (in << 4), lane);
      c[0][in] = wmma_f16(a0, b, c[0][in]);
      c[1][in] = wmma_f16(a1, b, c[1][in]);
    }
  };

  const int nk = K >> 5;     // >= 2 for all GEMMs in this model
  // prologue: stage 0 -> buffer 0
  stage_async<BN>(Ab, Wb, lda, ldw, bm, bn, 0, baseA, baseB, tid);

  for (int ks = 0; ks < nk - 1; ++ks) {
    const int nxt = (ks + 1) & 1;
    stage_async<BN>(Ab, Wb, lda, ldw, bm, bn, (ks + 1) << 5,
                    baseA + nxt * bufA, baseB + nxt * bufB, tid);
    // only the newest stage's 2+BN/32 copies may remain outstanding
    asm volatile("s_wait_asynccnt %0" :: "n"(2 + (BN >> 5)) : "memory");
    __syncthreads();                      // all waves' copies visible
    compute_stage(ks & 1);
    __syncthreads();                      // buf reused 2 stages later
  }
  asm volatile("s_wait_asynccnt 0" ::: "memory");
  __syncthreads();
  compute_stage((nk - 1) & 1);

  // Epilogue. C fragment: VGPR r, lanes 0-15 -> M=r, lanes 16-31 -> M=r+8.
#pragma unroll
  for (int im = 0; im < 2; ++im) {
#pragma unroll
    for (int in = 0; in < NF; ++in) {
#pragma unroll
      for (int r = 0; r < 8; ++r) {
        int m = bm + wm + (im << 4) + r + ((lane >> 4) << 3);
        int n = bn + wn + (in << 4) + (lane & 15);
        float acc = c[im][in][r];
        if (MODE == EP_SCORES) {
          outh[(size_t)batch * strideO + (size_t)m * ldo + n] =
              (_Float16)(acc * INV_SCALE);
        } else if (MODE == EP_QKHEAD) {
          acc += bias[n];
          int b = m >> 11, t = m & (SEQ - 1), h = n >> 6, hd = n & 63;
          outh[(((size_t)(b * NHEAD + h) * SEQ) + t) * HEADDIM + hd] =
              (_Float16)acc;
        } else if (MODE == EP_VTRANS) {
          acc += bias[n];
          int b = m >> 11, t = m & (SEQ - 1), h = n >> 6, hd = n & 63;
          outh[((size_t)(b * NHEAD + h) * HEADDIM + hd) * SEQ + t] =
              (_Float16)acc;
        } else if (MODE == EP_ATTNO) {
          int b = batch >> 3, h = batch & 7;
          outh[((size_t)b * SEQ + m) * DMODEL + h * HEADDIM + n] =
              (_Float16)acc;
        } else if (MODE == EP_F16BR) {
          acc += bias[n];
          outh[(size_t)m * ldo + n] = (_Float16)fmaxf(acc, 0.0f);
        } else {  // EP_RESID
          float* p = outf + (size_t)m * ldo + n;
          *p = *p + acc + bias[n];
        }
      }
    }
  }
}

// ---------------- reductions ----------------
__device__ __forceinline__ float block_reduce_add(float v, float* red, int tid) {
  red[tid] = v; __syncthreads();
#pragma unroll
  for (int s = 128; s > 0; s >>= 1) {
    if (tid < s) red[tid] += red[tid + s];
    __syncthreads();
  }
  float r = red[0]; __syncthreads();
  return r;
}
__device__ __forceinline__ float block_reduce_max(float v, float* red, int tid) {
  red[tid] = v; __syncthreads();
#pragma unroll
  for (int s = 128; s > 0; s >>= 1) {
    if (tid < s) red[tid] = fmaxf(red[tid], red[tid + s]);
    __syncthreads();
  }
  float r = red[0]; __syncthreads();
  return r;
}

// ---------------- layernorm (row of 512), templated output type ------------
template <typename OT>
__global__ __launch_bounds__(256)
void layernorm_k(const float* __restrict__ x, const float* __restrict__ g,
                 const float* __restrict__ bta, OT* __restrict__ out) {
  __shared__ float red[256];
  int m = blockIdx.x, tid = threadIdx.x;
  const float* row = x + (size_t)m * DMODEL;
  float v0 = row[tid], v1 = row[tid + 256];
  float mean = block_reduce_add(v0 + v1, red, tid) * (1.0f / DMODEL);
  float d0 = v0 - mean, d1 = v1 - mean;
  float var = block_reduce_add(d0 * d0 + d1 * d1, red, tid) * (1.0f / DMODEL);
  float rstd = rsqrtf(var + LNEPS);
  OT* orow = out + (size_t)m * DMODEL;
  orow[tid]       = (OT)(d0 * rstd * g[tid] + bta[tid]);
  orow[tid + 256] = (OT)(d1 * rstd * g[tid + 256] + bta[tid + 256]);
}

// ---------------- softmax over all heads of one (b,t) row + head-average ---
__global__ __launch_bounds__(256)
void softmax_avg_k(_Float16* __restrict__ scores, float* __restrict__ amap,
                   int layer) {
  __shared__ float red[256];
  int t = blockIdx.x, b = blockIdx.y, tid = threadIdx.x;
  float avg[8] = {0, 0, 0, 0, 0, 0, 0, 0};
#pragma unroll 1
  for (int h = 0; h < NHEAD; ++h) {
    _Float16* row = scores + ((size_t)(b * NHEAD + h) * SEQ + t) * SEQ;
    float v[8], mx = -1e30f;
#pragma unroll
    for (int j = 0; j < 8; ++j) {
      v[j] = (float)row[tid + (j << 8)];
      mx = fmaxf(mx, v[j]);
    }
    mx = block_reduce_max(mx, red, tid);
    float sum = 0.0f;
#pragma unroll
    for (int j = 0; j < 8; ++j) { v[j] = __expf(v[j] - mx); sum += v[j]; }
    sum = block_reduce_add(sum, red, tid);
    float inv = 1.0f / sum;
#pragma unroll
    for (int j = 0; j < 8; ++j) {
      float p = v[j] * inv;
      row[tid + (j << 8)] = (_Float16)p;   // probs written back in place
      avg[j] += p;
    }
  }
  float* orow = amap + (((size_t)layer * BATCH + b) * SEQ + t) * SEQ;
#pragma unroll
  for (int j = 0; j < 8; ++j) orow[tid + (j << 8)] = avg[j] * (1.0f / NHEAD);
}

// ---------------- misc elementwise kernels ----------------
__global__ __launch_bounds__(256)
void embed_k(const int* __restrict__ x, const float* __restrict__ emb,
             const float* __restrict__ pos, float* __restrict__ h) {
  size_t i = (size_t)blockIdx.x * 256 + threadIdx.x;   // over ROWS*DMODEL
  int m = (int)(i >> 9), cidx = (int)(i & 511);
  int t = m & (SEQ - 1);
  h[i] = emb[(size_t)x[m] * DMODEL + cidx] + pos[(size_t)t * DMODEL + cidx];
}

__global__ __launch_bounds__(256)
void f32tof16_k(const float* __restrict__ a, _Float16* __restrict__ o, int n) {
  int i = blockIdx.x * 256 + threadIdx.x;
  if (i < n) o[i] = (_Float16)a[i];
}

__global__ __launch_bounds__(256)
void col_mean_k(const float* __restrict__ x, float* __restrict__ out) {
  __shared__ float red[256];
  int cidx = blockIdx.x, b = blockIdx.y, tid = threadIdx.x;
  float s = 0.0f;
  for (int t = tid; t < SEQ; t += 256)
    s += x[((size_t)b * SEQ + t) * DMODEL + cidx];
  s = block_reduce_add(s, red, tid);
  if (tid == 0) out[b * DMODEL + cidx] = s * (1.0f / SEQ);
}

// ---------------- host orchestration ----------------
extern "C" void kernel_launch(void* const* d_in, const int* in_sizes, int n_in,
                              void* d_out, int out_size, void* d_ws,
                              size_t ws_size, hipStream_t stream) {
  (void)in_sizes; (void)n_in; (void)out_size; (void)ws_size;
  const int*   x     = (const int*)d_in[0];
  const float* emb   = (const float*)d_in[1];
  const float* pos   = (const float*)d_in[2];
  const float* Wq    = (const float*)d_in[3];
  const float* bq    = (const float*)d_in[4];
  const float* Wk    = (const float*)d_in[5];
  const float* bk    = (const float*)d_in[6];
  const float* Wv    = (const float*)d_in[7];
  const float* bv    = (const float*)d_in[8];
  const float* Wo    = (const float*)d_in[9];
  const float* bo    = (const float*)d_in[10];
  const float* W1    = (const float*)d_in[11];
  const float* b1    = (const float*)d_in[12];
  const float* W2    = (const float*)d_in[13];
  const float* b2    = (const float*)d_in[14];
  const float* ln1g  = (const float*)d_in[15];
  const float* ln1b  = (const float*)d_in[16];
  const float* ln2g  = (const float*)d_in[17];
  const float* ln2b  = (const float*)d_in[18];
  const float* lnfg  = (const float*)d_in[19];
  const float* lnfb  = (const float*)d_in[20];

  // Workspace carve-out (bump allocator, 256B aligned)
  char* p = (char*)d_ws;
  auto carve = [&](size_t bytes) -> char* {
    char* r = p;
    p += (bytes + 255) & ~(size_t)255;
    return r;
  };
  const size_t NW_SMALL = (size_t)NLAYER * DMODEL * DMODEL;  // 1,048,576
  const size_t NW_BIG   = (size_t)NLAYER * DFF * DMODEL;     // 4,194,304
  _Float16* wq16 = (_Float16*)carve(NW_SMALL * 2);
  _Float16* wk16 = (_Float16*)carve(NW_SMALL * 2);
  _Float16* wv16 = (_Float16*)carve(NW_SMALL * 2);
  _Float16* wo16 = (_Float16*)carve(NW_SMALL * 2);
  _Float16* w116 = (_Float16*)carve(NW_BIG * 2);
  _Float16* w216 = (_Float16*)carve(NW_BIG * 2);
  float*    hbuf  = (float*)carve((size_t)ROWS * DMODEL * 4);
  _Float16* xnorm = (_Float16*)carve((size_t)ROWS * DMODEL * 2);
  _Float16* q16   = (_Float16*)carve((size_t)ROWS * DMODEL * 2);
  _Float16* k16   = (_Float16*)carve((size_t)ROWS * DMODEL * 2);
  _Float16* vt16  = (_Float16*)carve((size_t)ROWS * DMODEL * 2);
  _Float16* at16  = (_Float16*)carve((size_t)ROWS * DMODEL * 2);
  _Float16* ff16  = (_Float16*)carve((size_t)ROWS * DFF * 2);
  _Float16* sc16  = (_Float16*)carve((size_t)BATCH * NHEAD * SEQ * SEQ * 2);
  float* lnbuf = (float*)sc16;  // reuse probs region for final LN (8MB << 128MB)

  float* out_mean = (float*)d_out;                        // [B, DMODEL]
  float* amap     = (float*)d_out + BATCH * DMODEL;       // [L,B,T,T]

  // 1) weights -> f16
  f32tof16_k<<<(int)((NW_SMALL + 255) / 256), 256, 0, stream>>>(Wq, wq16, (int)NW_SMALL);
  f32tof16_k<<<(int)((NW_SMALL + 255) / 256), 256, 0, stream>>>(Wk, wk16, (int)NW_SMALL);
  f32tof16_k<<<(int)((NW_SMALL + 255) / 256), 256, 0, stream>>>(Wv, wv16, (int)NW_SMALL);
  f32tof16_k<<<(int)((NW_SMALL + 255) / 256), 256, 0, stream>>>(Wo, wo16, (int)NW_SMALL);
  f32tof16_k<<<(int)((NW_BIG   + 255) / 256), 256, 0, stream>>>(W1, w116, (int)NW_BIG);
  f32tof16_k<<<(int)((NW_BIG   + 255) / 256), 256, 0, stream>>>(W2, w216, (int)NW_BIG);

  // 2) embedding gather
  embed_k<<<(ROWS * DMODEL) / 256, 256, 0, stream>>>(x, emb, pos, hbuf);

  const dim3 blk(128);
  for (int l = 0; l < NLAYER; ++l) {
    const _Float16* wql = wq16 + (size_t)l * DMODEL * DMODEL;
    const _Float16* wkl = wk16 + (size_t)l * DMODEL * DMODEL;
    const _Float16* wvl = wv16 + (size_t)l * DMODEL * DMODEL;
    const _Float16* wol = wo16 + (size_t)l * DMODEL * DMODEL;
    const _Float16* w1l = w116 + (size_t)l * DFF * DMODEL;
    const _Float16* w2l = w216 + (size_t)l * DMODEL * DFF;

    // LN1
    layernorm_k<_Float16><<<ROWS, 256, 0, stream>>>(
        hbuf, ln1g + l * DMODEL, ln1b + l * DMODEL, xnorm);

    // Q, K -> [B,H,T,64] ; V -> [B,H,64,T]
    gemm_wmma<EP_QKHEAD, 128><<<dim3(DMODEL / 128, ROWS / 64, 1), blk, 0, stream>>>(
        xnorm, wql, bq + l * DMODEL, q16, nullptr,
        ROWS, DMODEL, DMODEL, DMODEL, DMODEL, 0, 0, 0, 0);
    gemm_wmma<EP_QKHEAD, 128><<<dim3(DMODEL / 128, ROWS / 64, 1), blk, 0, stream>>>(
        xnorm, wkl, bk + l * DMODEL, k16, nullptr,
        ROWS, DMODEL, DMODEL, DMODEL, DMODEL, 0, 0, 0, 0);
    gemm_wmma<EP_VTRANS, 128><<<dim3(DMODEL / 128, ROWS / 64, 1), blk, 0, stream>>>(
        xnorm, wvl, bv + l * DMODEL, vt16, nullptr,
        ROWS, DMODEL, DMODEL, DMODEL, DMODEL, 0, 0, 0, 0);

    // scores = Q K^T / 8, batched over (b,h)
    gemm_wmma<EP_SCORES, 128><<<dim3(SEQ / 128, SEQ / 64, BATCH * NHEAD), blk, 0, stream>>>(
        q16, k16, nullptr, sc16, nullptr,
        SEQ, SEQ, HEADDIM, HEADDIM, HEADDIM, SEQ,
        (long long)SEQ * HEADDIM, (long long)SEQ * HEADDIM,
        (long long)SEQ * SEQ);

    // softmax per row over all heads, write probs in place + head-avg map
    softmax_avg_k<<<dim3(SEQ, BATCH), 256, 0, stream>>>(sc16, amap, l);

    // attn = probs @ V  (batched over (b,h)) -> [B,T,DMODEL]
    gemm_wmma<EP_ATTNO, 64><<<dim3(HEADDIM / 64, SEQ / 64, BATCH * NHEAD), blk, 0, stream>>>(
        sc16, vt16, nullptr, at16, nullptr,
        SEQ, HEADDIM, SEQ, SEQ, SEQ, 0,
        (long long)SEQ * SEQ, (long long)HEADDIM * SEQ, 0);

    // O projection + residual (in-place f32 accumulate into hbuf)
    gemm_wmma<EP_RESID, 128><<<dim3(DMODEL / 128, ROWS / 64, 1), blk, 0, stream>>>(
        at16, wol, bo + l * DMODEL, nullptr, hbuf,
        ROWS, DMODEL, DMODEL, DMODEL, DMODEL, DMODEL, 0, 0, 0);

    // LN2
    layernorm_k<_Float16><<<ROWS, 256, 0, stream>>>(
        hbuf, ln2g + l * DMODEL, ln2b + l * DMODEL, xnorm);

    // FFN
    gemm_wmma<EP_F16BR, 128><<<dim3(DFF / 128, ROWS / 64, 1), blk, 0, stream>>>(
        xnorm, w1l, b1 + l * DFF, ff16, nullptr,
        ROWS, DFF, DMODEL, DMODEL, DMODEL, DFF, 0, 0, 0);
    gemm_wmma<EP_RESID, 128><<<dim3(DMODEL / 128, ROWS / 64, 1), blk, 0, stream>>>(
        ff16, w2l, b2 + l * DMODEL, nullptr, hbuf,
        ROWS, DMODEL, DFF, DFF, DFF, DMODEL, 0, 0, 0);
  }

  // final LN + mean over T
  layernorm_k<float><<<ROWS, 256, 0, stream>>>(hbuf, lnfg, lnfb, lnbuf);
  col_mean_k<<<dim3(DMODEL, BATCH), 256, 0, stream>>>(lnbuf, out_mean);
}
